// SCAN_60954175865300
// MI455X (gfx1250) — matrix operand order repaired and checked by reading, not compiled, not run on hardware
//
#include <hip/hip_runtime.h>
#include <hip/hip_bf16.h>

// ---------------------------------------------------------------------------
// SCAN similarity, fused CDNA5 kernel (double-buffered WMMA pipeline).
//   S[c,i,:,:] = img_emb[i] (36x1024) @ cap_emb[c]^T (48x1024)  via bf16 WMMA
//   t2i: leaky -> l2norm over w -> softmax over r -> <.,S>/48
//   i2t: leaky -> l2norm over r -> softmax over w -> <.,S>/36
//   (g = cap_mean.img_mean cancels exactly under the ra/sum renormalization)
// ---------------------------------------------------------------------------

typedef __attribute__((ext_vector_type(16))) __bf16        v16bf;
typedef __attribute__((ext_vector_type(4)))  __bf16        v4bf;
typedef __attribute__((ext_vector_type(8)))  float         v8f;
typedef __attribute__((ext_vector_type(4)))  float         f32x4;
typedef __attribute__((ext_vector_type(4)))  unsigned int  u32x4;

union Frag { u32x4 u[2]; v16bf v; };

#define LAM 9.0f
#define EPS 1e-8f

constexpr int N_IMG = 128, N_CAP = 128, N_REG = 36, N_WRD = 48, DIM = 1024;
constexpr int TI      = 4;      // images per block (cap tile reused TI times)
constexpr int THREADS = 288;    // 9 waves: 3x3 grid of 16x16 WMMA tiles
constexpr int KC      = 128;    // k-chunk for img staging
constexpr int NCHUNK  = DIM / KC;          // 8
constexpr int NCK     = TI * NCHUNK;       // 32 chunks per block
constexpr int KSTEPS  = KC / 32;           // 4 WMMAs per chunk
constexpr int CAP_LD  = 1032;   // bf16 row stride (16B align + bank spread)
constexpr int IMG_LD  = 136;    // bf16 chunk row stride
constexpr int S_LD    = 49;     // f32 S-tile stride (odd -> conflict-free cols)

constexpr int CAP_ELEMS = 48 * CAP_LD;
constexpr int IMG_ELEMS = 48 * IMG_LD;
constexpr int S_ELEMS   = 48 * S_LD;
constexpr size_t LDS_BYTES =
    (size_t)CAP_ELEMS * 2 + 2 * (size_t)IMG_ELEMS * 2 +
    (size_t)(S_ELEMS + 36 + 48 + 48 + 36) * 4;   // ~132 KB (<320 KB WGP LDS)

// per-thread staging: 36*128 floats / 288 threads = 4 x float4
constexpr int STG = 4;

__device__ __forceinline__ float lrelu(float x) { return x > 0.f ? x : 0.1f * x; }

__device__ __forceinline__ float wave_sum(float v) {
#pragma unroll
  for (int o = 16; o; o >>= 1) v += __shfl_xor(v, o, 32);
  return v;
}
__device__ __forceinline__ float wave_max(float v) {
#pragma unroll
  for (int o = 16; o; o >>= 1) v = fmaxf(v, __shfl_xor(v, o, 32));
  return v;
}

__global__ void __launch_bounds__(THREADS)
scan_kernel(const float* __restrict__ imgE,   // [128,36,1024]
            const float* __restrict__ capE,   // [128,48,1024]
            float* __restrict__ out)          // [128(i),128(c)]
{
  extern __shared__ char smem[];
  __bf16* lds_cap  = (__bf16*)smem;                    // 48 x CAP_LD
  __bf16* lds_img0 = lds_cap + CAP_ELEMS;              // ping
  __bf16* lds_img1 = lds_img0 + IMG_ELEMS;             // pong
  float*  lds_S    = (float*)(lds_img1 + IMG_ELEMS);   // 48 x S_LD
  float*  rownorm  = lds_S + S_ELEMS;                  // 36
  float*  colnorm  = rownorm + 36;                     // 48
  float*  colacc   = colnorm + 48;                     // 48
  float*  rowacc   = colacc + 48;                      // 36

  const int tid  = threadIdx.x;
  const int wv   = tid >> 5;
  const int lane = tid & 31;
  const int half = lane >> 4;     // 16-lane group
  const int l16  = lane & 15;
  const int mt   = wv / 3;        // output tile row (M)
  const int nt   = wv % 3;        // output tile col (N)

  const int c  = blockIdx.x;
  const int i0 = blockIdx.y * TI;

  // ---- stage caption rows (fp32 -> bf16), float4-vectorized, once ----
  for (int idx = tid; idx < 48 * DIM / 4; idx += THREADS) {
    int r = idx >> 8, k4 = idx & 255;
    f32x4 f = *(const f32x4*)(capE + ((size_t)c * 48 + r) * DIM + k4 * 4);
    v4bf b = { (__bf16)f[0], (__bf16)f[1], (__bf16)f[2], (__bf16)f[3] };
    *(v4bf*)(lds_cap + r * CAP_LD + k4 * 4) = b;
  }
  // ---- zero the M-padding rows (36..47) of both img buffers, once ----
  for (int idx = tid; idx < 12 * IMG_LD; idx += THREADS) {
    lds_img0[36 * IMG_LD + idx] = (__bf16)0.0f;
    lds_img1[36 * IMG_LD + idx] = (__bf16)0.0f;
  }
  // ---- prologue: stage chunk ck=0 into ping buffer ----
#pragma unroll
  for (int j = 0; j < STG; ++j) {
    int idx = tid + j * THREADS;
    int r = idx >> 5, k4 = idx & 31;
    f32x4 f = *(const f32x4*)(imgE + ((size_t)i0 * N_REG + r) * DIM + k4 * 4);
    v4bf b = { (__bf16)f[0], (__bf16)f[1], (__bf16)f[2], (__bf16)f[3] };
    *(v4bf*)(lds_img0 + r * IMG_LD + k4 * 4) = b;
  }

  for (int ii = 0; ii < TI; ++ii) {
    const int img = i0 + ii;
    v8f acc = {};

    // ============ WMMA pipeline: S = A(36x1024) @ B(48x1024)^T ============
    for (int kc = 0; kc < NCHUNK; ++kc) {
      const int ck = ii * NCHUNK + kc;          // global chunk counter
      __syncthreads();                          // stage(ck) visible; buf free

      // -- issue global loads for chunk ck+1 (next image's chunk 0 at kc=7) --
      const int  nck  = ck + 1;
      const bool have = nck < NCK;
      f32x4 st[STG];
      if (have) {
        const int nimg = i0 + (nck >> 3);
        const int nkc  = nck & 7;
#pragma unroll
        for (int j = 0; j < STG; ++j) {
          int idx = tid + j * THREADS;
          int r = idx >> 5, k4 = idx & 31;
          st[j] = *(const f32x4*)(imgE + ((size_t)nimg * N_REG + r) * DIM +
                                  nkc * KC + k4 * 4);
        }
      }

      // -- batch-load ALL fragments for this chunk, then fire WMMAs
      //    back-to-back (one dscnt wait, XDL pipe stays busy) --
      const __bf16* buf = (ck & 1) ? lds_img1 : lds_img0;
      // A frag: row m, K = kk + 8*half + {0..7, 16..23}
      const __bf16* arow = buf + (mt * 16 + l16) * IMG_LD + half * 8;
      // B frag: col n, K = kk + 8*half + {0..7, 16..23}
      const __bf16* brow = lds_cap + (nt * 16 + l16) * CAP_LD + kc * KC + half * 8;
      Frag Af[KSTEPS], Bf[KSTEPS];
#pragma unroll
      for (int t = 0; t < KSTEPS; ++t) {
        const int kk = t * 32;
        Af[t].u[0] = *(const u32x4*)(arow + kk);       // K = kk..kk+7   (+8*half)
        Af[t].u[1] = *(const u32x4*)(arow + kk + 16);  // K = kk+16..+23 (+8*half)
        Bf[t].u[0] = *(const u32x4*)(brow + kk);       // K = kk..kk+7   (+8*half)
        Bf[t].u[1] = *(const u32x4*)(brow + kk + 16);  // K = kk+16..+23 (+8*half)
      }
#pragma unroll
      for (int t = 0; t < KSTEPS; ++t)
        acc = __builtin_amdgcn_wmma_f32_16x16x32_bf16(
            false, Af[t].v, false, Bf[t].v, (short)0, acc, false, false);

      // -- convert + store chunk ck+1 into the other buffer --
      if (have) {
        __bf16* dst = (nck & 1) ? lds_img1 : lds_img0;
#pragma unroll
        for (int j = 0; j < STG; ++j) {
          int idx = tid + j * THREADS;
          int r = idx >> 5, k4 = idx & 31;
          v4bf b = { (__bf16)st[j][0], (__bf16)st[j][1],
                     (__bf16)st[j][2], (__bf16)st[j][3] };
          *(v4bf*)(dst + r * IMG_LD + k4 * 4) = b;
        }
      }
    }

    // ---- spill S tile to LDS (C/D layout: lane=N, vgpr v -> M = v+8*half) ----
    {
      const int colw = nt * 16 + l16;
#pragma unroll
      for (int v = 0; v < 8; ++v) {
        int row = mt * 16 + half * 8 + v;
        lds_S[row * S_LD + colw] = acc[v];
      }
    }
    __syncthreads();

    // ================= attention math on the 36x48 tile ====================
    // Pass 1a: per-row l2 norms of leaky(S)      (t2i, norm over w)
    for (int r = wv; r < N_REG; r += 9) {
      float v0 = lrelu(lds_S[r * S_LD + lane]);
      float v1 = (lane < 16) ? lrelu(lds_S[r * S_LD + 32 + lane]) : 0.0f;
      float s = wave_sum(v0 * v0 + v1 * v1);
      if (lane == 0) rownorm[r] = sqrtf(s + EPS) + EPS;
    }
    // Pass 1b: per-col l2 norms of leaky(S)      (i2t, norm over r)
    for (int w = wv; w < N_WRD; w += 9) {
      float v0 = lrelu(lds_S[lane * S_LD + w]);   // r = lane (<36)
      float v1 = (lane < 4) ? lrelu(lds_S[(32 + lane) * S_LD + w]) : 0.0f;
      float s = wave_sum(v0 * v0 + v1 * v1);
      if (lane == 0) colnorm[w] = sqrtf(s + EPS) + EPS;
    }
    __syncthreads();

    // Pass 2a: t2i — softmax over r per column, contract with S
    for (int w = wv; w < N_WRD; w += 9) {
      float S0 = lds_S[lane * S_LD + w];
      float x0 = LAM * lrelu(S0) / rownorm[lane];
      bool ok1 = lane < 4;
      float S1 = ok1 ? lds_S[(32 + lane) * S_LD + w] : 0.0f;
      float x1 = ok1 ? LAM * lrelu(S1) / rownorm[32 + lane] : -1e30f;
      float m  = wave_max(fmaxf(x0, x1));
      float e0 = __expf(x0 - m);
      float e1 = ok1 ? __expf(x1 - m) : 0.0f;
      float Z   = wave_sum(e0 + e1);
      float num = wave_sum(e0 * S0 + e1 * S1);
      if (lane == 0) colacc[w] = num / Z;
    }
    // Pass 2b: i2t — softmax over w per row, contract with S
    for (int r = wv; r < N_REG; r += 9) {
      float S0 = lds_S[r * S_LD + lane];
      float x0 = LAM * lrelu(S0) / colnorm[lane];
      bool ok1 = lane < 16;
      float S1 = ok1 ? lds_S[r * S_LD + 32 + lane] : 0.0f;
      float x1 = ok1 ? LAM * lrelu(S1) / colnorm[32 + lane] : -1e30f;
      float m  = wave_max(fmaxf(x0, x1));
      float e0 = __expf(x0 - m);
      float e1 = ok1 ? __expf(x1 - m) : 0.0f;
      float Z   = wave_sum(e0 + e1);
      float num = wave_sum(e0 * S0 + e1 * S1);
      if (lane == 0) rowacc[r] = num / Z;
    }
    __syncthreads();

    // ---- final reduce: out[i,c] = mean_w colacc + mean_r rowacc ----
    if (wv == 0) {
      float t = colacc[lane] + ((lane < 16) ? colacc[32 + lane] : 0.0f);
      float u = rowacc[lane] + ((lane < 4) ? rowacc[32 + lane] : 0.0f);
      float tsum = wave_sum(t);
      float usum = wave_sum(u);
      if (lane == 0)
        out[(size_t)img * N_CAP + c] = tsum / (float)N_WRD + usum / (float)N_REG;
    }
    // next image's first __syncthreads() fences wave0's reads above
  }
}

extern "C" void kernel_launch(void* const* d_in, const int* in_sizes, int n_in,
                              void* d_out, int out_size, void* d_ws, size_t ws_size,
                              hipStream_t stream) {
  (void)in_sizes; (void)n_in; (void)out_size; (void)d_ws; (void)ws_size;
  const float* imgE = (const float*)d_in[0];   // img_emb  [128,36,1024]
  // d_in[1] = img_mean, d_in[3] = cap_mean: mathematically cancel (g / sum_r g)
  const float* capE = (const float*)d_in[2];   // cap_emb  [128,48,1024]
  float* out = (float*)d_out;                  // [128, 128] = (t2i+i2t).T

  dim3 grid(N_CAP, N_IMG / TI);   // 128 x 32 blocks
  dim3 block(THREADS);            // 9 waves
  scan_kernel<<<grid, block, LDS_BYTES, stream>>>(imgE, capE, out);
}
